// PhdLoss_8040178778855
// MI455X (gfx1250) — matrix operand order repaired and checked by reading, not compiled, not run on hardware
//
#include <hip/hip_runtime.h>
#include <math.h>

typedef __attribute__((ext_vector_type(16))) _Float16 v16h;
typedef __attribute__((ext_vector_type(8)))  _Float16 v8h;
typedef __attribute__((ext_vector_type(8)))  float    v8f;
typedef __attribute__((ext_vector_type(4)))  float    v4f;

#define NTRACK 128
#define S_FR   32
#define DFEAT  256
#define NVEC   (NTRACK * S_FR)      // 4096
#define NBLK   (NTRACK * NTRACK)    // 16384
#define EPSF   1e-12f
#define MARGINF 0.3f
#define K_AP_C 3
#define K_AN_C 6

// ---------------------------------------------------------------------------
// Kernel 1: L2-normalize each 256-d vector, emit f16 copy + f32 sum-of-squares
// of the *converted* values (so distances stay consistent with the f16 gram).
// One wave (32 lanes) per vector, 8 floats per lane.
// ---------------------------------------------------------------------------
__global__ __launch_bounds__(256) void phd_normalize(const float* __restrict__ in,
                                                     _Float16* __restrict__ Xh,
                                                     float* __restrict__ sq) {
  int lane = threadIdx.x & 31;
  int wave = threadIdx.x >> 5;
  int vec  = blockIdx.x * 8 + wave;            // 512 blocks * 8 waves = 4096
  const float* p = in + (size_t)vec * DFEAT + lane * 8;
  v4f x0 = *(const v4f*)p;
  v4f x1 = *(const v4f*)(p + 4);
  float ss = x0.x*x0.x + x0.y*x0.y + x0.z*x0.z + x0.w*x0.w
           + x1.x*x1.x + x1.y*x1.y + x1.z*x1.z + x1.w*x1.w;
  #pragma unroll
  for (int off = 16; off; off >>= 1) ss += __shfl_xor(ss, off, 32);
  float inv = 1.0f / fmaxf(sqrtf(ss), EPSF);

  float xv[8] = {x0.x, x0.y, x0.z, x0.w, x1.x, x1.y, x1.z, x1.w};
  v8h h;
  float hs = 0.0f;
  #pragma unroll
  for (int e = 0; e < 8; ++e) {
    _Float16 hv = (_Float16)(xv[e] * inv);
    h[e] = hv;
    float f = (float)hv;
    hs += f * f;
  }
  #pragma unroll
  for (int off = 16; off; off >>= 1) hs += __shfl_xor(hs, off, 32);

  *(v8h*)(Xh + (size_t)vec * DFEAT + lane * 8) = h;
  if (lane == 0) sq[vec] = hs;
}

// ---------------------------------------------------------------------------
// k-th largest of 32 per-lane values via iterative wave-max extraction.
// Ballot + ffs removes exactly one instance per round (duplicate-safe).
// ---------------------------------------------------------------------------
__device__ __forceinline__ float kth_largest32(float v, int k) {
  int lane = threadIdx.x & 31;
  float kth = -1e30f;
  for (int t = 0; t < k; ++t) {
    float m = v;
    #pragma unroll
    for (int off = 16; off; off >>= 1) m = fmaxf(m, __shfl_xor(m, off, 32));
    kth = m;
    unsigned long long ball = __ballot(v == m);
    int first = __ffsll(ball) - 1;
    if (lane == first) v = -1e30f;
  }
  return kth;
}

// ---------------------------------------------------------------------------
// Kernel 2: one 32x32 tracklet-pair block per 32-thread workgroup.
// 2x2 tiles of v_wmma_f32_16x16x32_f16 over K=256 (8 k-steps, 32 WMMAs).
// Distances -> padded LDS tile -> row/col mins -> k-th-largest -> 1 scalar.
// ---------------------------------------------------------------------------
__global__ __launch_bounds__(32) void phd_blocks(const _Float16* __restrict__ Xh,
                                                 const float* __restrict__ sq,
                                                 const int* __restrict__ tgt,
                                                 float* __restrict__ val) {
  __shared__ float tile[32 * 33];              // pitch 33 -> conflict-free both ways
  int bid  = blockIdx.x;                       // 0..16383
  int ti   = bid >> 7;                         // tracklet i (rows)
  int tj   = bid & 127;                        // tracklet j (cols)
  int lane = threadIdx.x;
  int half = lane >> 4;                        // 0: lanes 0-15, 1: lanes 16-31
  int lr   = lane & 15;

  const _Float16* Ar = Xh + (size_t)ti * S_FR * DFEAT;
  const _Float16* Br = Xh + (size_t)tj * S_FR * DFEAT;

  v8f c00 = {}, c01 = {}, c10 = {}, c11 = {};

  #pragma unroll
  for (int kk = 0; kk < 8; ++kk) {
    int kb = kk * 32;
    union { v16h v; v8h h[2]; } a0, a1, b0, b1;
    // A 16x32 f16 frag: lane holds row lr, K chunks {half*8 .. +7} and {+16 .. +23}
    {
      const _Float16* r0 = Ar + (size_t)lr * DFEAT + kb + half * 8;
      a0.h[0] = *(const v8h*)(r0);
      a0.h[1] = *(const v8h*)(r0 + 16);
      const _Float16* r1 = Ar + (size_t)(16 + lr) * DFEAT + kb + half * 8;
      a1.h[0] = *(const v8h*)(r1);
      a1.h[1] = *(const v8h*)(r1 + 16);
    }
    // B 32x16 f16 frag: lane holds column lr (= row lr of Xj), K chunks {half*8 .. +7} and {+16 .. +23}
    {
      const _Float16* q0 = Br + (size_t)lr * DFEAT + kb + half * 8;
      b0.h[0] = *(const v8h*)(q0);
      b0.h[1] = *(const v8h*)(q0 + 16);
      const _Float16* q1 = Br + (size_t)(16 + lr) * DFEAT + kb + half * 8;
      b1.h[0] = *(const v8h*)(q1);
      b1.h[1] = *(const v8h*)(q1 + 16);
    }
    c00 = __builtin_amdgcn_wmma_f32_16x16x32_f16(false, a0.v, false, b0.v, (short)0, c00, false, false);
    c01 = __builtin_amdgcn_wmma_f32_16x16x32_f16(false, a0.v, false, b1.v, (short)0, c01, false, false);
    c10 = __builtin_amdgcn_wmma_f32_16x16x32_f16(false, a1.v, false, b0.v, (short)0, c10, false, false);
    c11 = __builtin_amdgcn_wmma_f32_16x16x32_f16(false, a1.v, false, b1.v, (short)0, c11, false, false);
  }

  // squared-norm side inputs
  float sqj0 = sq[tj * 32 + lr];               // col tile 0
  float sqj1 = sq[tj * 32 + 16 + lr];          // col tile 1
  const float* sqi = sq + ti * 32 + half * 8;  // rows m = mi*16 + half*8 + r
  float si0[8], si1[8];
  #pragma unroll
  for (int r = 0; r < 8; ++r) { si0[r] = sqi[r]; si1[r] = sqi[16 + r]; }

  // dist = sqrt(clip(sq_i + sq_j - 2*dot, EPS)) -> LDS (C frag: elem r = row
  // mi*16 + half*8 + r, column ni*16 + lr)
  #pragma unroll
  for (int r = 0; r < 8; ++r) {
    int m0 = half * 8 + r;
    int m1 = 16 + half * 8 + r;
    tile[m0 * 33 + lr]      = sqrtf(fmaxf(si0[r] + sqj0 - 2.0f * c00[r], EPSF));
    tile[m0 * 33 + 16 + lr] = sqrtf(fmaxf(si0[r] + sqj1 - 2.0f * c01[r], EPSF));
    tile[m1 * 33 + lr]      = sqrtf(fmaxf(si1[r] + sqj0 - 2.0f * c10[r], EPSF));
    tile[m1 * 33 + 16 + lr] = sqrtf(fmaxf(si1[r] + sqj1 - 2.0f * c11[r], EPSF));
  }
  __syncthreads();

  // dij[lane] = column min, dji[lane] = row min
  float cmin = 1e30f, rmin = 1e30f;
  #pragma unroll 4
  for (int m = 0; m < 32; ++m) cmin = fminf(cmin, tile[m * 33 + lane]);
  #pragma unroll 4
  for (int n = 0; n < 32; ++n) rmin = fminf(rmin, tile[lane * 33 + n]);

  bool pos = (tgt[ti] == tgt[tj]);
  int k = pos ? K_AP_C : K_AN_C;
  float kij = kth_largest32(cmin, k);
  float kji = kth_largest32(rmin, k);
  if (lane == 0) {
    val[bid] = pos ? fmaxf(kij, kji) : fminf(kij, kji);
  }
}

// ---------------------------------------------------------------------------
// Kernel 3: per-anchor max-over-positives / min-over-negatives + mean ReLU.
// ---------------------------------------------------------------------------
__global__ __launch_bounds__(128) void phd_final(const float* __restrict__ val,
                                                 const int* __restrict__ tgt,
                                                 float* __restrict__ out) {
  __shared__ float red[128];
  int i = threadIdx.x;
  int ti = tgt[i];
  float ap = -1e30f, an = 1e30f;
  for (int j = 0; j < 128; ++j) {
    float v = val[i * 128 + j];
    if (tgt[j] == ti) ap = fmaxf(ap, v);
    else              an = fminf(an, v);
  }
  red[i] = fmaxf(ap - an + MARGINF, 0.0f);
  __syncthreads();
  for (int off = 64; off; off >>= 1) {
    if (i < off) red[i] += red[i + off];
    __syncthreads();
  }
  if (i == 0) out[0] = red[0] * (1.0f / 128.0f);
}

// ---------------------------------------------------------------------------
extern "C" void kernel_launch(void* const* d_in, const int* in_sizes, int n_in,
                              void* d_out, int out_size, void* d_ws, size_t ws_size,
                              hipStream_t stream) {
  const float* inputs  = (const float*)d_in[0];   // (128, 32, 256) f32
  const int*   targets = (const int*)d_in[1];     // (128,) i32
  float* out = (float*)d_out;

  char* ws = (char*)d_ws;
  _Float16* Xh = (_Float16*)ws;                           // 4096*256 f16 = 2 MB
  float* sq  = (float*)(ws + (size_t)NVEC * DFEAT * 2);   // 4096 f32
  float* val = sq + NVEC;                                 // 16384 f32

  phd_normalize<<<NVEC / 8, 256, 0, stream>>>(inputs, Xh, sq);
  phd_blocks<<<NBLK, 32, 0, stream>>>(Xh, sq, targets, val);
  phd_final<<<1, 128, 0, stream>>>(val, targets, out);
}
